// ALiBiMultiHeadAttention_34256659153053
// MI455X (gfx1250) — matrix operand order repaired and checked by reading, not compiled, not run on hardware
//
#include <hip/hip_runtime.h>

// ---------------------------------------------------------------------------
// ALiBi causal MHA for MI455X (gfx1250, wave32, WMMA).
// B=2 S=2048 Dm=1024 H=16 D=64.  bf16 WMMA operands, f32 accumulate.
// K1: qkv GEMM -> q,k [B,H,S,D] bf16 ; v transposed [B,H,D,S] bf16
// K2: fused flash attention (per-wave 16-query tile, online softmax)
// K3: out GEMM + bias -> f32
// Software pipelined: tile k+1 global loads in flight during tile k WMMAs.
// ---------------------------------------------------------------------------

typedef __attribute__((ext_vector_type(16))) __bf16 v16bf;
typedef __attribute__((ext_vector_type(8)))  __bf16 v8bf;
typedef __attribute__((ext_vector_type(4)))  __bf16 v4bf;
typedef __attribute__((ext_vector_type(8)))  float  v8f;

#define KPAD 40   // 32 bf16 of data + pad -> 80B rows, 16B aligned fragments

__device__ __forceinline__ v16bf cat8(v8bf lo, v8bf hi) {
  return __builtin_shufflevector(lo, hi, 0,1,2,3,4,5,6,7,8,9,10,11,12,13,14,15);
}
__device__ __forceinline__ v8f wmma_bf16(v16bf a, v16bf b, v8f c) {
  // D = A(16x32) * B(32x16) + C, f32 accumulate
  return __builtin_amdgcn_wmma_f32_16x16x32_bf16(false, a, false, b,
                                                 (short)0, c, false, false);
}

// ---------------------------------------------------------------------------
// Kernel 1: C[4096,3072] = x[4096,1024] @ Wqkv[1024,3072] + bqkv
// 128x128 block tile, K-step 32, 8 waves (2x4), wave tile 64x32.
// Epilogue scatters into q/k ([B,H,S,D] bf16) and vT ([B,H,D,S] bf16).
// ---------------------------------------------------------------------------
__global__ __launch_bounds__(256) void qkv_gemm(
    const float* __restrict__ x, const float* __restrict__ W,
    const float* __restrict__ bias,
    unsigned short* __restrict__ qb_, unsigned short* __restrict__ kb_,
    unsigned short* __restrict__ vtb_)
{
  __bf16* qb  = reinterpret_cast<__bf16*>(qb_);
  __bf16* kb  = reinterpret_cast<__bf16*>(kb_);
  __bf16* vtb = reinterpret_cast<__bf16*>(vtb_);

  __shared__ __bf16 As[128][KPAD];   // [M][K] bf16
  __shared__ __bf16 Bs[128][KPAD];   // [N][K] bf16 (pre-transposed)

  const int t    = threadIdx.x;
  const int lane = t & 31, w = t >> 5;
  const int hi   = lane >> 4, l16 = lane & 15;
  const int m0   = blockIdx.y * 128, n0 = blockIdx.x * 128;
  const int wm   = (w & 1) * 64,     wn = (w >> 1) * 32;
  const int gB   = t & 31;           // B: n-group of 4 cols
  const int krB  = (t >> 5) * 4;     // B: 4 consecutive k-rows

  float4 ra[4], rb[4];
  // issue all global loads for one K-tile (no data use -> loads overlap)
  auto load_tiles = [&](int k0) {
    #pragma unroll
    for (int i = 0; i < 4; ++i) {              // A: 128x32 f32, coalesced
      const int f = t + i * 256, row = f >> 3, c4 = f & 7;
      ra[i] = *(const float4*)(x + (size_t)(m0 + row) * 1024 + k0 + c4 * 4);
    }
    #pragma unroll
    for (int j = 0; j < 4; ++j)                // B: 4 k-rows x 4 n-cols
      rb[j] = *(const float4*)(W + (size_t)(k0 + krB + j) * 3072 + n0 + gB * 4);
  };
  // convert + store staged registers into LDS
  auto store_tiles = [&]() {
    #pragma unroll
    for (int i = 0; i < 4; ++i) {
      const int f = t + i * 256, row = f >> 3, c4 = f & 7;
      v4bf p; p[0] = (__bf16)ra[i].x; p[1] = (__bf16)ra[i].y;
              p[2] = (__bf16)ra[i].z; p[3] = (__bf16)ra[i].w;
      *(v4bf*)&As[row][c4 * 4] = p;            // 8B store
    }
    #pragma unroll
    for (int cidx = 0; cidx < 4; ++cidx) {     // transpose in registers
      v4bf p;
      #pragma unroll
      for (int j = 0; j < 4; ++j)
        p[j] = (__bf16)((const float*)&rb[j])[cidx];
      *(v4bf*)&Bs[gB * 4 + cidx][krB] = p;     // 8B store (krB 4-aligned)
    }
  };

  v8f acc[4][2];
  #pragma unroll
  for (int mt = 0; mt < 4; ++mt)
    #pragma unroll
    for (int nt = 0; nt < 2; ++nt)
      #pragma unroll
      for (int i = 0; i < 8; ++i) acc[mt][nt][i] = 0.f;

  load_tiles(0);
  store_tiles();
  __syncthreads();

  for (int k0 = 0; k0 < 1024; k0 += 32) {
    const bool more = (k0 + 32) < 1024;
    if (more) load_tiles(k0 + 32);             // in flight during WMMAs

    #pragma unroll
    for (int mt = 0; mt < 4; ++mt) {
      const int row = wm + mt * 16 + l16;
      v16bf a = cat8(*(const v8bf*)&As[row][hi * 8],
                     *(const v8bf*)&As[row][16 + hi * 8]);
      #pragma unroll
      for (int nt = 0; nt < 2; ++nt) {
        const int col = wn + nt * 16 + l16;
        v16bf b = cat8(*(const v8bf*)&Bs[col][hi * 8],
                       *(const v8bf*)&Bs[col][16 + hi * 8]);
        acc[mt][nt] = wmma_bf16(a, b, acc[mt][nt]);
      }
    }
    __syncthreads();                           // all waves done reading LDS
    if (more) {
      store_tiles();                           // wait_loadcnt lands here
      __syncthreads();
    }
  }

  // ---- epilogue: bias + scatter to q / k / vT ------------------------------
  #pragma unroll
  for (int mt = 0; mt < 4; ++mt)
    #pragma unroll
    for (int nt = 0; nt < 2; ++nt) {
      const int n_g = n0 + wn + nt * 16 + l16;
      const float bv = bias[n_g];
      const int which = n_g >> 10;             // 0=q 1=k 2=v
      const int hh = (n_g >> 6) & 15, d = n_g & 63;
      #pragma unroll
      for (int r = 0; r < 8; ++r) {
        const int m_g = m0 + wm + mt * 16 + r + hi * 8;
        const int bb = m_g >> 11, s = m_g & 2047;
        const __bf16 o = (__bf16)(acc[mt][nt][r] + bv);
        const long rowmaj = ((long)(bb * 16 + hh) * 2048 + s) * 64 + d;
        if (which == 0)      qb[rowmaj] = o;
        else if (which == 1) kb[rowmaj] = o;
        else vtb[((long)(bb * 16 + hh) * 64 + d) * 2048 + s] = o;
      }
    }
}

// ---------------------------------------------------------------------------
// Kernel 2: fused flash attention with ALiBi + causal + pad mask.
// One wave = one 16-query tile of one (b,h).  32-key chunks:
//   scores  : 2 key-tiles x 2 WMMA (D=64)  -> C-layout 16x16 tiles
//   softmax : per-row stats via shfl_xor across 16-lane halves (wave32)
//   P @ V   : P bounced through LDS (C-layout -> A-fragment), 4 WMMA (D=64)
// ---------------------------------------------------------------------------
__global__ __launch_bounds__(256) void attn_kernel(
    const unsigned short* __restrict__ qb_, const unsigned short* __restrict__ kb_,
    const unsigned short* __restrict__ vtb_, const int* __restrict__ amask,
    unsigned short* __restrict__ aout_)
{
  const __bf16* qb  = reinterpret_cast<const __bf16*>(qb_);
  const __bf16* kb  = reinterpret_cast<const __bf16*>(kb_);
  const __bf16* vtb = reinterpret_cast<const __bf16*>(vtb_);
  __bf16* aout = reinterpret_cast<__bf16*>(aout_);

  __shared__ __bf16 Pl[8][16][KPAD];           // per-wave P transpose buffer

  const int t = threadIdx.x, lane = t & 31, w = t >> 5;
  const int hi = lane >> 4, l16 = lane & 15;
  const int h = blockIdx.y, b = blockIdx.z;
  const int bh = b * 16 + h;
  const int q0 = blockIdx.x * 128 + w * 16;

  // Q A-fragments (K-dim = head dim, two 32-wide ksteps)
  const __bf16* qrow = qb + ((long)bh * 2048 + q0 + l16) * 64;
  const v16bf aq0 = cat8(*(const v8bf*)(qrow + hi * 8),
                         *(const v8bf*)(qrow + 16 + hi * 8));
  const v16bf aq1 = cat8(*(const v8bf*)(qrow + 32 + hi * 8),
                         *(const v8bf*)(qrow + 48 + hi * 8));

  const float slope = __builtin_exp2f(-0.5f * (float)(h + 1)); // ALiBi slope
  const float inv_scale = 0.125f;                              // 1/sqrt(64)

  float mrow[8], lrow[8];
  v8f acc[4];
  #pragma unroll
  for (int r = 0; r < 8; ++r) { mrow[r] = -1e30f; lrow[r] = 0.f; }
  #pragma unroll
  for (int dt = 0; dt < 4; ++dt)
    #pragma unroll
    for (int i = 0; i < 8; ++i) acc[dt][i] = 0.f;

  const int nch = ((q0 + 15) >> 5) + 1;        // causal chunk bound
  for (int c = 0; c < nch; ++c) {
    const int jc = c * 32;

    if (c + 1 < nch) {                         // global_prefetch_b8 next chunk
      __builtin_prefetch(kb + ((long)bh * 2048 + jc + 32 + lane) * 64, 0, 0);
      __builtin_prefetch(vtb + ((long)bh * 64 + lane) * 2048 + jc + 32, 0, 0);
    }

    // ---- scores: Q @ K^T for two 16-key tiles ------------------------------
    v8f sc[2];
    #pragma unroll
    for (int kt = 0; kt < 2; ++kt) {
      const int j = jc + kt * 16 + l16;
      const __bf16* krow = kb + ((long)bh * 2048 + j) * 64;
      v16bf bk0 = cat8(*(const v8bf*)(krow + hi * 8),
                       *(const v8bf*)(krow + 16 + hi * 8));
      v16bf bk1 = cat8(*(const v8bf*)(krow + 32 + hi * 8),
                       *(const v8bf*)(krow + 48 + hi * 8));
      v8f ct;
      #pragma unroll
      for (int i = 0; i < 8; ++i) ct[i] = 0.f;
      ct = wmma_bf16(aq0, bk0, ct);
      ct = wmma_bf16(aq1, bk1, ct);
      sc[kt] = ct;
    }
    const int mv0 = amask[b * 2048 + jc + l16];
    const int mv1 = amask[b * 2048 + jc + 16 + l16];

    // ---- online softmax ----------------------------------------------------
    #pragma unroll
    for (int r = 0; r < 8; ++r) {
      const int qg = q0 + r + hi * 8;
      float p0, p1, best;
      {
        const int j0 = jc + l16, j1 = jc + 16 + l16;
        p0 = sc[0][r] * inv_scale + slope * (float)(j0 - qg);
        p1 = sc[1][r] * inv_scale + slope * (float)(j1 - qg);
        if (j0 > qg || mv0 == 0) p0 = -1e30f;
        if (j1 > qg || mv1 == 0) p1 = -1e30f;
        best = fmaxf(p0, p1);
      }
      #pragma unroll
      for (int xm = 1; xm < 16; xm <<= 1)      // reduce within 16-lane half
        best = fmaxf(best, __shfl_xor(best, xm));
      const float mnew  = fmaxf(mrow[r], best);
      const float alpha = __expf(mrow[r] - mnew);
      mrow[r] = mnew;
      p0 = __expf(p0 - mnew);
      p1 = __expf(p1 - mnew);
      float ps = p0 + p1;
      #pragma unroll
      for (int xm = 1; xm < 16; xm <<= 1)
        ps += __shfl_xor(ps, xm);
      lrow[r] = lrow[r] * alpha + ps;
      #pragma unroll
      for (int dt = 0; dt < 4; ++dt) acc[dt][r] *= alpha;
      // C-layout -> LDS [q][j] for A-fragment re-read
      Pl[w][r + hi * 8][l16]      = (__bf16)p0;
      Pl[w][r + hi * 8][16 + l16] = (__bf16)p1;
    }
    asm volatile("s_wait_dscnt 0" ::: "memory");  // per-wave LDS visibility

    // ---- P @ V over this 32-key chunk (K-dim = keys) -----------------------
    const __bf16* prow = &Pl[w][l16][0];
    const v16bf ap = cat8(*(const v8bf*)(prow + hi * 8),
                          *(const v8bf*)(prow + 16 + hi * 8));
    #pragma unroll
    for (int dt = 0; dt < 4; ++dt) {
      const __bf16* vrow =
          vtb + ((long)bh * 64 + dt * 16 + l16) * 2048 + jc;
      const v16bf bv = cat8(*(const v8bf*)(vrow + hi * 8),
                            *(const v8bf*)(vrow + 16 + hi * 8));
      acc[dt] = wmma_bf16(ap, bv, acc[dt]);
    }
  }

  // ---- normalize + store [B,S,Dm] bf16 ------------------------------------
  #pragma unroll
  for (int r = 0; r < 8; ++r) {
    const float inv = lrow[r] > 0.f ? 1.f / lrow[r] : 0.f;
    const long row = ((long)b * 2048 + q0 + r + hi * 8) * 1024 + h * 64;
    #pragma unroll
    for (int dt = 0; dt < 4; ++dt)
      aout[row + dt * 16 + l16] = (__bf16)(acc[dt][r] * inv);
  }
}

// ---------------------------------------------------------------------------
// Kernel 3: out[4096,1024] = attn[4096,1024](bf16) @ Wout[1024,1024] + bout
// ---------------------------------------------------------------------------
__global__ __launch_bounds__(256) void out_gemm(
    const unsigned short* __restrict__ A_, const float* __restrict__ W,
    const float* __restrict__ bias, float* __restrict__ out)
{
  const __bf16* A = reinterpret_cast<const __bf16*>(A_);

  __shared__ __bf16 As[128][KPAD];
  __shared__ __bf16 Bs[128][KPAD];

  const int t    = threadIdx.x;
  const int lane = t & 31, w = t >> 5;
  const int hi   = lane >> 4, l16 = lane & 15;
  const int m0   = blockIdx.y * 128, n0 = blockIdx.x * 128;
  const int wm   = (w & 1) * 64,     wn = (w >> 1) * 32;
  const int gB   = t & 31, krB = (t >> 5) * 4;

  v8bf rA[2];
  float4 rb[4];
  auto load_tiles = [&](int k0) {
    #pragma unroll
    for (int i = 0; i < 2; ++i) {              // A bf16: 512 8-elem units
      const int u = t + i * 256, row = u >> 2, c8 = u & 3;
      rA[i] = *(const v8bf*)(A + (size_t)(m0 + row) * 1024 + k0 + c8 * 8);
    }
    #pragma unroll
    for (int j = 0; j < 4; ++j)                // B: 4 k-rows x 4 n-cols
      rb[j] = *(const float4*)(W + (size_t)(k0 + krB + j) * 1024 + n0 + gB * 4);
  };
  auto store_tiles = [&]() {
    #pragma unroll
    for (int i = 0; i < 2; ++i) {
      const int u = t + i * 256, row = u >> 2, c8 = u & 3;
      *(v8bf*)&As[row][c8 * 8] = rA[i];        // 16B store
    }
    #pragma unroll
    for (int cidx = 0; cidx < 4; ++cidx) {
      v4bf p;
      #pragma unroll
      for (int j = 0; j < 4; ++j)
        p[j] = (__bf16)((const float*)&rb[j])[cidx];
      *(v4bf*)&Bs[gB * 4 + cidx][krB] = p;     // 8B store
    }
  };

  v8f acc[4][2];
  #pragma unroll
  for (int mt = 0; mt < 4; ++mt)
    #pragma unroll
    for (int nt = 0; nt < 2; ++nt)
      #pragma unroll
      for (int i = 0; i < 8; ++i) acc[mt][nt][i] = 0.f;

  load_tiles(0);
  store_tiles();
  __syncthreads();

  for (int k0 = 0; k0 < 1024; k0 += 32) {
    const bool more = (k0 + 32) < 1024;
    if (more) load_tiles(k0 + 32);

    #pragma unroll
    for (int mt = 0; mt < 4; ++mt) {
      const int row = wm + mt * 16 + l16;
      v16bf a = cat8(*(const v8bf*)&As[row][hi * 8],
                     *(const v8bf*)&As[row][16 + hi * 8]);
      #pragma unroll
      for (int nt = 0; nt < 2; ++nt) {
        const int col = wn + nt * 16 + l16;
        v16bf b = cat8(*(const v8bf*)&Bs[col][hi * 8],
                       *(const v8bf*)&Bs[col][16 + hi * 8]);
        acc[mt][nt] = wmma_bf16(a, b, acc[mt][nt]);
      }
    }
    __syncthreads();
    if (more) {
      store_tiles();
      __syncthreads();
    }
  }

  #pragma unroll
  for (int mt = 0; mt < 4; ++mt)
    #pragma unroll
    for (int nt = 0; nt < 2; ++nt) {
      const int n_g = n0 + wn + nt * 16 + l16;
      const float bv = bias[n_g];
      #pragma unroll
      for (int r = 0; r < 8; ++r) {
        const int m_g = m0 + wm + mt * 16 + r + hi * 8;
        out[(size_t)m_g * 1024 + n_g] = acc[mt][nt][r] + bv;
      }
    }
}

// ---------------------------------------------------------------------------
extern "C" void kernel_launch(void* const* d_in, const int* in_sizes, int n_in,
                              void* d_out, int out_size, void* d_ws, size_t ws_size,
                              hipStream_t stream) {
  (void)in_sizes; (void)n_in; (void)out_size; (void)ws_size;
  const float* x     = (const float*)d_in[0];
  const int*   amask = (const int*)d_in[1];
  const float* Wqkv  = (const float*)d_in[2];
  const float* bqkv  = (const float*)d_in[3];
  const float* Wout  = (const float*)d_in[4];
  const float* bout  = (const float*)d_in[5];
  float* out = (float*)d_out;

  // workspace: q 8MB | k 8MB | vT 8MB | attn_out 8MB (all bf16)
  char* ws = (char*)d_ws;
  unsigned short* qb   = (unsigned short*)(ws);
  unsigned short* kb   = (unsigned short*)(ws + (size_t)8  * 1024 * 1024);
  unsigned short* vtb  = (unsigned short*)(ws + (size_t)16 * 1024 * 1024);
  unsigned short* aout = (unsigned short*)(ws + (size_t)24 * 1024 * 1024);

  qkv_gemm  <<<dim3(24, 32),    256, 0, stream>>>(x, Wqkv, bqkv, qb, kb, vtb);
  attn_kernel<<<dim3(16, 16, 2), 256, 0, stream>>>(qb, kb, vtb, amask, aout);
  out_gemm  <<<dim3(8, 32),     256, 0, stream>>>(aout, Wout, bout, out);
}
